// GumbelAttention_51187420234398
// MI455X (gfx1250) — matrix operand order repaired and checked by reading, not compiled, not run on hardware
//
#include <hip/hip_runtime.h>
#include <hip/hip_bf16.h>
#include <math.h>

// ---------------------------------------------------------------------------
// GumbelAttention forward for MI455X (gfx1250), wave32.
// All GEMMs use V_WMMA_F32_16X16X32_BF16 with an f32->bf16 hi/lo split
// (bf16x3: hi*hi + hi*lo + lo*hi, f32 accumulate).  LDS tiles are pre-split
// into packed bf16 hi/lo planes (one u32 word = two K-consecutive bf16);
// plane row stride is 36 words (16B-aligned, bank-conflict-free) so A-side
// fragment loads lower to ds_load_b128.
// B=8, NQ=200 (n=100), S=4096, E=2048, H=8, D=256.
// ---------------------------------------------------------------------------

typedef __attribute__((ext_vector_type(16))) __bf16   v16bf;
typedef __attribute__((ext_vector_type(8)))  float    v8f;
typedef __attribute__((ext_vector_type(8)))  unsigned v8u;

struct Frag { v16bf hi, lo; };

__device__ __forceinline__ v8f zero8() {
  v8f z = {0.f, 0.f, 0.f, 0.f, 0.f, 0.f, 0.f, 0.f};
  return z;
}

// Pack bf16(a) into [15:0], bf16(b) into [31:16]  (v_cvt_pk_bf16_f32).
__device__ __forceinline__ unsigned pk_hi(float a, float b) {
  unsigned short ua = __builtin_bit_cast(unsigned short, (__bf16)a);
  unsigned short ub = __builtin_bit_cast(unsigned short, (__bf16)b);
  return (unsigned)ua | ((unsigned)ub << 16);
}
__device__ __forceinline__ float bf_rt(float a) { return (float)(__bf16)a; }
__device__ __forceinline__ unsigned pk_lo(float a, float b) {
  return pk_hi(a - bf_rt(a), b - bf_rt(b));
}

__device__ __forceinline__ v8u cat44(uint4 a, uint4 b) {
  v8u r;
  r[0] = a.x; r[1] = a.y; r[2] = a.z; r[3] = a.w;
  r[4] = b.x; r[5] = b.y; r[6] = b.z; r[7] = b.w;
  return r;
}

// --- fragment loaders from packed hi/lo planes ------------------------------
// A layout (16x32 bf16): element j -> K = j + 16*(lane>>4);
// pair words per lane: run {0..7} (+8*(lane>>4)).  Row stride
// is a multiple of 4 words -> two ds_load_b128 per plane.
__device__ __forceinline__ Frag frag_a(const unsigned* hi, const unsigned* lo,
                                       int ldw, int lane, int kw0) {
  const int off = (lane & 15) * ldw + kw0 + ((lane >> 4) << 3);
  const v8u hw = cat44(*(const uint4*)(hi + off), *(const uint4*)(hi + off + 4));
  const v8u lw = cat44(*(const uint4*)(lo + off), *(const uint4*)(lo + off + 4));
  Frag f;
  f.hi = __builtin_bit_cast(v16bf, hw);
  f.lo = __builtin_bit_cast(v16bf, lw);
  return f;
}

// B layout (32x16 bf16), natural [wordRow][col] planes (column strided):
// element j -> K = j + 16*(lane>>4); pair p -> wordRow kw0 + p + 8*(lane>>4).
__device__ __forceinline__ Frag frag_b(const unsigned* hi, const unsigned* lo,
                                       int ldw, int lane, int kw0) {
  const int off = (kw0 + ((lane >> 4) << 3)) * ldw + (lane & 15);
  v8u hw, lw;
#pragma unroll
  for (int p = 0; p < 8; ++p) {
    hw[p] = hi[off + p * ldw];
    lw[p] = lo[off + p * ldw];
  }
  Frag f;
  f.hi = __builtin_bit_cast(v16bf, hw);
  f.lo = __builtin_bit_cast(v16bf, lw);
  return f;
}

// B from transposed storage [nRow][word]: 8 contiguous words -> 2x b128.
__device__ __forceinline__ Frag frag_bt(const unsigned* hi, const unsigned* lo,
                                        int ldw, int lane, int kw0) {
  const int off = (lane & 15) * ldw + kw0 + ((lane >> 4) << 3);
  const v8u hw = cat44(*(const uint4*)(hi + off), *(const uint4*)(hi + off + 4));
  const v8u lw = cat44(*(const uint4*)(lo + off), *(const uint4*)(lo + off + 4));
  Frag f;
  f.hi = __builtin_bit_cast(v16bf, hw);
  f.lo = __builtin_bit_cast(v16bf, lw);
  return f;
}

__device__ __forceinline__ v8f wmma3(const Frag& a, const Frag& b, v8f c) {
  c = __builtin_amdgcn_wmma_f32_16x16x32_bf16(false, a.hi, false, b.hi,
                                              (short)0, c, false, false);
  c = __builtin_amdgcn_wmma_f32_16x16x32_bf16(false, a.hi, false, b.lo,
                                              (short)0, c, false, false);
  c = __builtin_amdgcn_wmma_f32_16x16x32_bf16(false, a.lo, false, b.hi,
                                              (short)0, c, false, false);
  return c;
}

// ---------------------------------------------------------------------------
// Generic C[M,N] = A[M,K] @ W[K,N] + bias.  Requires N%128==0, K%64==0.
// Block tile 64x128, 8 waves as 2(M) x 4(N), each wave 32x32; BK=64.
// ---------------------------------------------------------------------------
#define GB_BM 64
#define GB_BN 128
#define GB_BK 64
#define GB_KW 32   // word columns (K pairs)
#define GB_LDA 36  // A plane row stride (words, 16B aligned)
#define GB_LDB 132 // B plane row stride (words)

__global__ __launch_bounds__(256) void gemm_bias_kernel(
    const float* __restrict__ A, const float* __restrict__ W,
    const float* __restrict__ bias, float* __restrict__ C,
    int M, int N, int K) {
  __shared__ unsigned AhiS[GB_BM * GB_LDA];
  __shared__ unsigned AloS[GB_BM * GB_LDA];
  __shared__ unsigned BhiS[GB_KW * GB_LDB];
  __shared__ unsigned BloS[GB_KW * GB_LDB];
  const int t = threadIdx.x, lane = t & 31, wave = t >> 5;
  const int m0 = blockIdx.y * GB_BM;
  const int n0 = blockIdx.x * GB_BN;
  const int waveM = wave >> 2, waveN = wave & 3;

  v8f acc[2][2];
#pragma unroll
  for (int i = 0; i < 2; ++i)
#pragma unroll
    for (int j = 0; j < 2; ++j) acc[i][j] = zero8();

  for (int kt = 0; kt < K; kt += GB_BK) {
    // Stage A tile 64x64 into hi/lo planes (pairs along K within a row).
    for (int i = t; i < GB_BM * 16; i += 256) {
      const int r = i >> 4, c4 = (i & 15) << 2;
      const int gr = m0 + r;
      float4 v = make_float4(0.f, 0.f, 0.f, 0.f);
      if (gr < M) v = *(const float4*)&A[(size_t)gr * K + kt + c4];
      const int w = r * GB_LDA + (c4 >> 1);
      AhiS[w + 0] = pk_hi(v.x, v.y); AhiS[w + 1] = pk_hi(v.z, v.w);
      AloS[w + 0] = pk_lo(v.x, v.y); AloS[w + 1] = pk_lo(v.z, v.w);
    }
    // Stage B tile 64x128: pack K row pairs (2rp, 2rp+1) per word.
    for (int i = t; i < GB_KW * (GB_BN / 4); i += 256) {
      const int rp = i >> 5, c4 = (i & 31) << 2;
      const float* wp = &W[(size_t)(kt + 2 * rp) * N + n0 + c4];
      const float4 v0 = *(const float4*)wp;
      const float4 v1 = *(const float4*)(wp + N);
      const int w = rp * GB_LDB + c4;
      BhiS[w + 0] = pk_hi(v0.x, v1.x); BhiS[w + 1] = pk_hi(v0.y, v1.y);
      BhiS[w + 2] = pk_hi(v0.z, v1.z); BhiS[w + 3] = pk_hi(v0.w, v1.w);
      BloS[w + 0] = pk_lo(v0.x, v1.x); BloS[w + 1] = pk_lo(v0.y, v1.y);
      BloS[w + 2] = pk_lo(v0.z, v1.z); BloS[w + 3] = pk_lo(v0.w, v1.w);
    }
    if (kt + GB_BK < K) {  // prefetch next K tile (global_prefetch_b8)
      const int gr = m0 + (t >> 2);
      if (gr < M) __builtin_prefetch(&A[(size_t)gr * K + kt + GB_BK], 0, 0);
      __builtin_prefetch(&W[(size_t)(kt + GB_BK + (t >> 3)) * N + n0 + (t & 7) * 16], 0, 0);
    }
    __syncthreads();

#pragma unroll
    for (int ks = 0; ks < 2; ++ks) {
      const int kw0 = ks * 16;
      Frag a[2], b[2];
#pragma unroll
      for (int mi = 0; mi < 2; ++mi)
        a[mi] = frag_a(&AhiS[(waveM * 32 + mi * 16) * GB_LDA],
                       &AloS[(waveM * 32 + mi * 16) * GB_LDA], GB_LDA, lane, kw0);
#pragma unroll
      for (int ni = 0; ni < 2; ++ni)
        b[ni] = frag_b(&BhiS[waveN * 32 + ni * 16],
                       &BloS[waveN * 32 + ni * 16], GB_LDB, lane, kw0);
#pragma unroll
      for (int mi = 0; mi < 2; ++mi)
#pragma unroll
        for (int ni = 0; ni < 2; ++ni)
          acc[mi][ni] = wmma3(a[mi], b[ni], acc[mi][ni]);
    }
    __syncthreads();
  }

  const int kh = (lane >> 4) << 3;
#pragma unroll
  for (int mi = 0; mi < 2; ++mi) {
#pragma unroll
    for (int ni = 0; ni < 2; ++ni) {
      const int mT = m0 + waveM * 32 + mi * 16;
      const int col = n0 + waveN * 32 + ni * 16 + (lane & 15);
      const float bv = bias[col];
#pragma unroll
      for (int r = 0; r < 8; ++r) {
        const int row = mT + r + kh;
        if (row < M) C[(size_t)row * N + col] = acc[mi][ni][r] + bv;
      }
    }
  }
}

// ---------------------------------------------------------------------------
// Row inverse L2 norms: rows 0..799 -> q1 rows of qp; rows 800.. -> kp rows.
// ---------------------------------------------------------------------------
__global__ __launch_bounds__(256) void norms_kernel(
    const float* __restrict__ qp, const float* __restrict__ kp,
    float* __restrict__ rq, float* __restrict__ rk) {
  const int row = blockIdx.x, t = threadIdx.x;
  const float* src;
  float* dst;
  int didx;
  if (row < 800) {
    const int b = row / 100, q = row % 100;
    src = qp + ((size_t)b * 200 + q) * 2048;
    dst = rq; didx = row;
  } else {
    const int r = row - 800;
    src = kp + (size_t)r * 2048;
    dst = rk; didx = r;
  }
  float s = 0.f;
  for (int i = t; i < 2048; i += 256) { float x = src[i]; s += x * x; }
  __shared__ float red[256];
  red[t] = s; __syncthreads();
  for (int o = 128; o; o >>= 1) {
    if (t < o) red[t] += red[t + o];
    __syncthreads();
  }
  if (t == 0) dst[didx] = 1.0f / sqrtf(red[0]);
}

// ---------------------------------------------------------------------------
// Scores per (b,h): mode 0 -> S1[q,k] = q1.k1; mode 1 -> S2 = (q2.k)/16.
// M=112 padded (100 valid), N tile 128, K=256, BK=64.  K-matrix staged in
// transposed plane layout [key][cWord] and consumed via frag_bt.
// ---------------------------------------------------------------------------
#define SC_BN 128
#define SC_BK 64
#define SC_KW 32
#define SC_LDA 36

__global__ __launch_bounds__(256) void scores_kernel(
    const float* __restrict__ qp, const float* __restrict__ kp,
    const float* __restrict__ rq, const float* __restrict__ rk,
    float* __restrict__ s1, float* __restrict__ s2) {
  __shared__ unsigned AhiS[112 * SC_LDA];
  __shared__ unsigned AloS[112 * SC_LDA];
  __shared__ unsigned KhiS[SC_BN * SC_LDA];
  __shared__ unsigned KloS[SC_BN * SC_LDA];
  const int t = threadIdx.x, lane = t & 31, wave = t >> 5;
  const int bh = blockIdx.y, b = bh >> 3, h = bh & 7;
  const int mode = blockIdx.z;
  const int n0 = blockIdx.x * SC_BN;
  const float* Abase = qp + ((size_t)b * 200 + (mode ? 100 : 0)) * 2048 + h * 256;
  const float* Kbase = kp + ((size_t)(b * 4096 + n0)) * 2048 + h * 256;

  v8f acc[7];
#pragma unroll
  for (int i = 0; i < 7; ++i) acc[i] = zero8();

  for (int kt = 0; kt < 256; kt += SC_BK) {
    for (int i = t; i < 112 * 16; i += 256) {
      const int r = i >> 4, c4 = (i & 15) << 2;
      float4 v = make_float4(0.f, 0.f, 0.f, 0.f);
      if (r < 100) v = *(const float4*)&Abase[(size_t)r * 2048 + kt + c4];
      const int w = r * SC_LDA + (c4 >> 1);
      AhiS[w + 0] = pk_hi(v.x, v.y); AhiS[w + 1] = pk_hi(v.z, v.w);
      AloS[w + 0] = pk_lo(v.x, v.y); AloS[w + 1] = pk_lo(v.z, v.w);
    }
    for (int i = t; i < SC_BN * 16; i += 256) {
      const int r = i >> 4, c4 = (i & 15) << 2;
      const float4 v = *(const float4*)&Kbase[(size_t)r * 2048 + kt + c4];
      const int w = r * SC_LDA + (c4 >> 1);
      KhiS[w + 0] = pk_hi(v.x, v.y); KhiS[w + 1] = pk_hi(v.z, v.w);
      KloS[w + 0] = pk_lo(v.x, v.y); KloS[w + 1] = pk_lo(v.z, v.w);
    }
    __syncthreads();
#pragma unroll
    for (int ks = 0; ks < 2; ++ks) {
      const int kw0 = ks * 16;
      Frag bf = frag_bt(&KhiS[(wave * 16) * SC_LDA], &KloS[(wave * 16) * SC_LDA],
                        SC_LDA, lane, kw0);
#pragma unroll
      for (int mi = 0; mi < 7; ++mi) {
        Frag af = frag_a(&AhiS[(mi * 16) * SC_LDA], &AloS[(mi * 16) * SC_LDA],
                         SC_LDA, lane, kw0);
        acc[mi] = wmma3(af, bf, acc[mi]);
      }
    }
    __syncthreads();
  }

  const int kh = (lane >> 4) << 3;
  const int col = n0 + wave * 16 + (lane & 15);  // global key index
  const float rkv = rk[b * 4096 + col];
  float* dst = mode ? s2 : s1;
#pragma unroll
  for (int mi = 0; mi < 7; ++mi) {
#pragma unroll
    for (int r = 0; r < 8; ++r) {
      const int q = mi * 16 + r + kh;
      if (q < 100) {
        float val = acc[mi][r];
        val = mode ? (val * 0.0625f) : (val * rq[b * 100 + q] * rkv);
        dst[((size_t)bh * 100 + q) * 4096 + col] = val;
      }
    }
  }
}

// ---------------------------------------------------------------------------
// Softmax row stats for S2: per row (6400) max + sum(exp).
// ---------------------------------------------------------------------------
__global__ __launch_bounds__(256) void softmax_stats_kernel(
    const float* __restrict__ s2, float* __restrict__ m2,
    float* __restrict__ d2) {
  const int row = blockIdx.x, t = threadIdx.x;
  const float* src = s2 + (size_t)row * 4096;
  __shared__ float red[256];
  float mx = -INFINITY;
  for (int i = t; i < 4096; i += 256) mx = fmaxf(mx, src[i]);
  red[t] = mx; __syncthreads();
  for (int o = 128; o; o >>= 1) {
    if (t < o) red[t] = fmaxf(red[t], red[t + o]);
    __syncthreads();
  }
  const float mAll = red[0];
  __syncthreads();
  float s = 0.f;
  for (int i = t; i < 4096; i += 256) s += __expf(src[i] - mAll);
  red[t] = s; __syncthreads();
  for (int o = 128; o; o >>= 1) {
    if (t < o) red[t] += red[t + o];
    __syncthreads();
  }
  if (t == 0) { m2[row] = mAll; d2[row] = red[0]; }
}

// ---------------------------------------------------------------------------
// JAX threefry2x32 (key = (0,42)) Gumbel noise + argmax over q (axis=-2).
// ---------------------------------------------------------------------------
__device__ __forceinline__ unsigned rotl32(unsigned x, int r) {
  return (x << r) | (x >> (32 - r));
}

__device__ __forceinline__ void threefry2x32_42(unsigned x0, unsigned x1,
                                                unsigned& o0, unsigned& o1) {
  const unsigned k0 = 0u, k1 = 42u, k2 = k0 ^ k1 ^ 0x1BD11BDAu;
  x0 += k0; x1 += k1;
  const int RA[4] = {13, 15, 26, 6};
  const int RB[4] = {17, 29, 16, 24};
#pragma unroll
  for (int i = 0; i < 4; ++i) { x0 += x1; x1 = rotl32(x1, RA[i]); x1 ^= x0; }
  x0 += k1; x1 += k2 + 1u;
#pragma unroll
  for (int i = 0; i < 4; ++i) { x0 += x1; x1 = rotl32(x1, RB[i]); x1 ^= x0; }
  x0 += k2; x1 += k0 + 2u;
#pragma unroll
  for (int i = 0; i < 4; ++i) { x0 += x1; x1 = rotl32(x1, RA[i]); x1 ^= x0; }
  x0 += k0; x1 += k1 + 3u;
#pragma unroll
  for (int i = 0; i < 4; ++i) { x0 += x1; x1 = rotl32(x1, RB[i]); x1 ^= x0; }
  x0 += k1; x1 += k2 + 4u;
#pragma unroll
  for (int i = 0; i < 4; ++i) { x0 += x1; x1 = rotl32(x1, RA[i]); x1 ^= x0; }
  x0 += k2; x1 += k0 + 5u;
  o0 = x0; o1 = x1;
}

__global__ __launch_bounds__(256) void gumbel_argmax_kernel(
    const float* __restrict__ s1, const float* __restrict__ tau_p,
    int* __restrict__ idxbuf) {
  const int bh = blockIdx.y;
  const int k = blockIdx.x * 256 + threadIdx.x;
  const unsigned half = 13107200u;  // (8*8*100*4096)/2
  const float tau = tau_p[0];
  float best = -INFINITY;
  int bi = 0;
  for (int q = 0; q < 100; ++q) {
    const size_t flat = ((size_t)bh * 100 + q) * 4096 + k;
    const unsigned uf = (unsigned)flat;
    const bool low = uf < half;
    const unsigned c0 = low ? uf : (uf - half);
    unsigned x0, x1;
    threefry2x32_42(c0, c0 + half, x0, x1);
    const unsigned bits = low ? x0 : x1;
    float u = __uint_as_float(0x3f800000u | (bits >> 9)) - 1.0f;
    u = u * (1.0f - 2e-6f) + 1e-6f;
    u = fmaxf(1e-6f, u);
    const float g = -logf(-logf(u));
    const float val = (s1[flat] + g) / tau;
    if (val > best) { best = val; bi = q; }
  }
  idxbuf[(size_t)bh * 4096 + k] = bi;
}

// ---------------------------------------------------------------------------
// out1: per (b,h,q): gather sum of vh rows whose argmax index == q.
// Index chunks staged to LDS via GLOBAL_LOAD_ASYNC_TO_LDS_B32 (ASYNCcnt).
// ---------------------------------------------------------------------------
__global__ __launch_bounds__(256) void out1_kernel(
    const float* __restrict__ vp, const int* __restrict__ idxbuf,
    float* __restrict__ out_pre) {
  const int bh = blockIdx.y, q = blockIdx.x;
  const int b = bh >> 3, h = bh & 7;
  const int t = threadIdx.x;
  __shared__ int sidx[256];
  const int* idx = idxbuf + (size_t)bh * 4096;
  const float* vbase = vp + (size_t)b * 4096 * 2048 + h * 256 + t;
  const unsigned lds_off = (unsigned)(t * 4);  // sidx is the only LDS object
  float acc = 0.f;
  for (int kt = 0; kt < 4096; kt += 256) {
    // async copy idx[kt+t] -> sidx[t]  (LDS address taken from the vdst VGPR)
    const int* gp = idx + kt + t;
    asm volatile("global_load_async_to_lds_b32 %0, %1, off"
                 :: "v"(lds_off), "v"(gp) : "memory");
    asm volatile("s_wait_asynccnt 0x0" ::: "memory");
    __syncthreads();
#pragma unroll 4
    for (int kk = 0; kk < 256; ++kk) {
      if (sidx[kk] == q) acc += vbase[(size_t)(kt + kk) * 2048];
    }
    __syncthreads();
  }
  out_pre[((size_t)b * 200 + q) * 2048 + h * 256 + t] = acc;
}

// ---------------------------------------------------------------------------
// out2: per (b,h): softmax(S2) @ vh via WMMA.  M=112(100), N tile 128,
// K=4096, BK=64.  exp applied while packing P planes; 1/rowsum at store.
// ---------------------------------------------------------------------------
__global__ __launch_bounds__(256) void out2_kernel(
    const float* __restrict__ s2, const float* __restrict__ vp,
    const float* __restrict__ m2, const float* __restrict__ d2,
    float* __restrict__ out_pre) {
  __shared__ unsigned PhiS[112 * SC_LDA];
  __shared__ unsigned PloS[112 * SC_LDA];
  __shared__ unsigned VhiS[SC_KW * GB_LDB];
  __shared__ unsigned VloS[SC_KW * GB_LDB];
  const int t = threadIdx.x, lane = t & 31, wave = t >> 5;
  const int bh = blockIdx.y, b = bh >> 3, h = bh & 7;
  const int c0 = blockIdx.x * SC_BN;
  const float* S2 = s2 + (size_t)bh * 100 * 4096;
  const float* M2 = m2 + bh * 100;

  v8f acc[7];
#pragma unroll
  for (int i = 0; i < 7; ++i) acc[i] = zero8();

  for (int kt = 0; kt < 4096; kt += SC_BK) {
    for (int i = t; i < 112 * 16; i += 256) {
      const int r = i >> 4, c4 = (i & 15) << 2;
      float4 v = make_float4(0.f, 0.f, 0.f, 0.f);
      if (r < 100) {
        const float4 sv = *(const float4*)&S2[(size_t)r * 4096 + kt + c4];
        const float mm = M2[r];
        v.x = __expf(sv.x - mm); v.y = __expf(sv.y - mm);
        v.z = __expf(sv.z - mm); v.w = __expf(sv.w - mm);
      }
      const int w = r * SC_LDA + (c4 >> 1);
      PhiS[w + 0] = pk_hi(v.x, v.y); PhiS[w + 1] = pk_hi(v.z, v.w);
      PloS[w + 0] = pk_lo(v.x, v.y); PloS[w + 1] = pk_lo(v.z, v.w);
    }
    for (int i = t; i < SC_KW * (SC_BN / 4); i += 256) {
      const int rp = i >> 5, c4 = (i & 31) << 2;
      const float* wp = &vp[((size_t)(b * 4096) + kt + 2 * rp) * 2048 +
                            h * 256 + c0 + c4];
      const float4 v0 = *(const float4*)wp;
      const float4 v1 = *(const float4*)(wp + 2048);
      const int w = rp * GB_LDB + c4;
      VhiS[w + 0] = pk_hi(v0.x, v1.x); VhiS[w + 1] = pk_hi(v0.y, v1.y);
      VhiS[w + 2] = pk_hi(v0.z, v1.z); VhiS[w + 3] = pk_hi(v0.w, v1.w);
      VloS[w + 0] = pk_lo(v0.x, v1.x); VloS[w + 1] = pk_lo(v0.y, v1.y);
      VloS[w + 2] = pk_lo(v0.z, v1.z); VloS[w + 3] = pk_lo(v0.w, v1.w);
    }
    __syncthreads();
#pragma unroll
    for (int ks = 0; ks < 2; ++ks) {
      const int kw0 = ks * 16;
      Frag bf = frag_b(&VhiS[wave * 16], &VloS[wave * 16], GB_LDB, lane, kw0);
#pragma unroll
      for (int mi = 0; mi < 7; ++mi) {
        Frag af = frag_a(&PhiS[(mi * 16) * SC_LDA], &PloS[(mi * 16) * SC_LDA],
                         SC_LDA, lane, kw0);
        acc[mi] = wmma3(af, bf, acc[mi]);
      }
    }
    __syncthreads();
  }

  const int kh = (lane >> 4) << 3;
  const int col = c0 + wave * 16 + (lane & 15);
#pragma unroll
  for (int mi = 0; mi < 7; ++mi) {
#pragma unroll
    for (int r = 0; r < 8; ++r) {
      const int q = mi * 16 + r + kh;
      if (q < 100) {
        out_pre[((size_t)b * 200 + 100 + q) * 2048 + h * 256 + col] =
            acc[mi][r] / d2[bh * 100 + q];
      }
    }
  }
}

// ---------------------------------------------------------------------------
// Host launch.  Workspace (floats): qp 3.28M | kp 67.1M | vp 67.1M |
// s1 26.2M | s2 26.2M | out_pre 3.28M | rq 800 | rk 32768 | m2 6400 |
// d2 6400 | idx 262144(int)  ~= 774 MB total.
// ---------------------------------------------------------------------------
extern "C" void kernel_launch(void* const* d_in, const int* in_sizes, int n_in,
                              void* d_out, int out_size, void* d_ws,
                              size_t ws_size, hipStream_t stream) {
  const float* q  = (const float*)d_in[0];
  const float* k  = (const float*)d_in[1];
  const float* v  = (const float*)d_in[2];
  const float* wq = (const float*)d_in[3];
  const float* bq = (const float*)d_in[4];
  const float* wk = (const float*)d_in[5];
  const float* bk = (const float*)d_in[6];
  const float* wv = (const float*)d_in[7];
  const float* bv = (const float*)d_in[8];
  const float* wo = (const float*)d_in[9];
  const float* bo = (const float*)d_in[10];
  const float* tau = (const float*)d_in[11];
  float* out = (float*)d_out;
  float* ws  = (float*)d_ws;

  const size_t QP_LEN = 1600ull * 2048;
  const size_t KP_LEN = 32768ull * 2048;
  const size_t S_LEN  = 64ull * 100 * 4096;

  float* qp      = ws;
  float* kp      = qp + QP_LEN;
  float* vp      = kp + KP_LEN;
  float* s1      = vp + KP_LEN;
  float* s2      = s1 + S_LEN;
  float* out_pre = s2 + S_LEN;
  float* rq      = out_pre + QP_LEN;
  float* rk      = rq + 800;
  float* m2      = rk + 32768;
  float* d2      = m2 + 6400;
  int*   idxb    = (int*)(d2 + 6400);

  dim3 blk(256);

  gemm_bias_kernel<<<dim3(16, 25),  blk, 0, stream>>>(q, wq, bq, qp, 1600, 2048, 2048);
  gemm_bias_kernel<<<dim3(16, 512), blk, 0, stream>>>(k, wk, bk, kp, 32768, 2048, 2048);
  gemm_bias_kernel<<<dim3(16, 512), blk, 0, stream>>>(v, wv, bv, vp, 32768, 2048, 2048);

  norms_kernel<<<800 + 32768, blk, 0, stream>>>(qp, kp, rq, rk);

  scores_kernel<<<dim3(32, 64, 2), blk, 0, stream>>>(qp, kp, rq, rk, s1, s2);

  softmax_stats_kernel<<<6400, blk, 0, stream>>>(s2, m2, d2);

  gumbel_argmax_kernel<<<dim3(16, 64), blk, 0, stream>>>(s1, tau, idxb);

  out1_kernel<<<dim3(100, 64), blk, 0, stream>>>(vp, idxb, out_pre);

  out2_kernel<<<dim3(2, 64), blk, 0, stream>>>(s2, vp, m2, d2, out_pre);

  gemm_bias_kernel<<<dim3(16, 25), blk, 0, stream>>>(out_pre, wo, bo, out, 1600, 2048, 2048);
}
